// MixedReservoir_65944927863155
// MI455X (gfx1250) — matrix operand order repaired and checked by reading, not compiled, not run on hardware
//
#include <hip/hip_runtime.h>

typedef __bf16 bf16t;
typedef bf16t v16bf __attribute__((ext_vector_type(16)));
typedef bf16t v8bf  __attribute__((ext_vector_type(8)));
typedef float v8f   __attribute__((ext_vector_type(8)));

#define T_   2048
#define IN_  128
#define L_   512
#define OUT_ 128
#define XPAD 8     // +16B row pad: 16 rows sweep all 64 LDS banks
#define APAD 8

__device__ __forceinline__ v16bf cat8(v8bf lo, v8bf hi) {
  return __builtin_shufflevector(lo, hi, 0,1,2,3,4,5,6,7,8,9,10,11,12,13,14,15);
}

// A fragment (16x32 bf16) at constant element offset `ofs` from per-phase base.
// lanes 0-15: K {k..k+7, k+16..k+23}; lanes 16-31: K {k+8..k+15, k+24..k+31}
__device__ __forceinline__ v16bf a_frag(const bf16t* ap, int ofs) {
  return cat8(*(const v8bf*)(ap + ofs), *(const v8bf*)(ap + ofs + 16));
}
// B fragment (32x16 bf16): lanes 0-15 col n K {k..k+7, k+16..k+23}; lanes 16-31 col n K {k+8..k+15, k+24..k+31}
__device__ __forceinline__ v16bf b_frag(const bf16t* bp, int ofs) {
  return cat8(*(const v8bf*)(bp + ofs), *(const v8bf*)(bp + ofs + 16));
}

// One matmul phase: acc += A(16xK from LDS) * B(KxN tile from global).
// Single base pointer per phase -> constant immediate offsets; 2-deep pipeline.
template<int KITERS>
__device__ __forceinline__ v8f mm_phase(const bf16t* Albs, int astride,
                                        const bf16t* __restrict__ Bt, int Kdim,
                                        int lane, int nbase, int kbeg, int inv, v8f acc) {
  const bf16t* ap = Albs + (lane & 15) * astride + kbeg + ((lane >> 4) << 3);
  const bf16t* bp = Bt + (size_t)(nbase + (lane & 15)) * Kdim
                       + kbeg + ((lane >> 4) << 3) + inv;
  v16bf aA = a_frag(ap, 0);
  v16bf bA = b_frag(bp, 0);
  v16bf aB = aA, bB = bA;
  if (KITERS > 1) { aB = a_frag(ap, 32); bB = b_frag(bp, 32); }
#pragma unroll
  for (int i = 0; i < KITERS; ++i) {
    v16bf aC = aB, bC = bB;
    if (i + 2 < KITERS) {                      // fragments i+2 issued before WMMA i
      aC = a_frag(ap, (i + 2) * 32);
      bC = b_frag(bp, (i + 2) * 32);
    }
    acc = __builtin_amdgcn_wmma_f32_16x16x32_bf16(false, aA, false, bA, (short)0, acc,
                                                  false, false);
    aA = aB; bA = bB; aB = aC; bB = bC;
  }
  return acc;
}

__device__ __forceinline__ float mixed_act(float v, int grp) {
  switch (grp) {
    case 0:  return __sinf(v);
    case 1:  return tanhf(v);
    case 2:  return __logf(v * v);
    case 3:  return fmaxf(v, 0.f);
    default: return 1.f / (1.f + __expf(-v));
  }
}

// CDNA5 async store: LDS -> global, tracked by ASYNCcnt (not STOREcnt), so no
// storecnt ack wait lands on the per-step barriers.
__device__ __forceinline__ void async_store_f32(float* gaddr, const float* lds) {
  unsigned ldso = (unsigned)(uintptr_t)lds;    // low 32 bits of generic ptr = LDS offset
  asm volatile("global_store_async_from_lds_b32 %0, %1, off"
               :: "v"(gaddr), "v"(ldso) : "memory");
}

__global__ void transpose_to_bf16(const float* __restrict__ in, bf16t* __restrict__ outp,
                                  int K, int N) {
  int idx = blockIdx.x * blockDim.x + threadIdx.x;
  if (idx < K * N) {
    int n = idx / K, k = idx - n * K;
    outp[idx] = (bf16t)in[k * N + n];     // out[n][k] = in[k][n]
  }
}

__global__ void convert_bf16(const float* __restrict__ in, bf16t* __restrict__ outp, int count) {
  int idx = blockIdx.x * blockDim.x + threadIdx.x;
  if (idx < count) outp[idx] = (bf16t)in[idx];
}

__global__ __launch_bounds__(1024, 1)
void reservoir_kernel(const float* __restrict__ x,
                      const float* __restrict__ a0,
                      const float* __restrict__ a1,
                      const float* __restrict__ bias,
                      const bf16t* __restrict__ t0t,   // [512][128]
                      const bf16t* __restrict__ r0t,   // [512][512]
                      const bf16t* __restrict__ t1t,   // [512][512]
                      const bf16t* __restrict__ r1t,   // [512][512]
                      const bf16t* __restrict__ Wb,    // [128][512] (W is already [N][K])
                      float* __restrict__ out)
{
  __shared__ __align__(16) bf16t bufX[16][IN_ + XPAD];
  __shared__ __align__(16) bf16t bufA[16][L_ + APAD];
  __shared__ __align__(16) bf16t bufB[16][L_ + APAD];
  __shared__ __align__(16) float part[4][16][OUT_];
  __shared__ __align__(16) float outStage[4][2048];   // async-store ring, 4 slots

  const int tid  = threadIdx.x;
  const int lane = tid & 31;
  const int w    = tid >> 5;             // wave 0..31, owns N-tile [16w, 16w+16)
  const int bt   = blockIdx.x;           // batch tile: rows 16*bt .. 16*bt+15

  const int ncol  = (w << 4) + (lane & 15);
  const int mrow  = (lane >> 4) << 3;    // C layout: lanes>=16 hold rows M=8..15
  const float alpha0 = a0[ncol];
  const float alpha1 = a1[ncol];
  const int grp = (w < 8) ? 0 : (w < 16) ? 1 : (w < 24) ? 2 : (w < 28) ? 3 : 4;

  const int rntile = w & 7;              // readout: N tile 0..7
  const int rkc    = w >> 3;             // readout: K chunk 0..3

  // x prefetch lanes: thread covers elements tid and tid+1024 of the 16x128 tile
  const int xm0 = tid >> 7, xc0 = tid & 127;
  const int xm1 = xm0 + 8;
  const float* xp0 = x + (size_t)(bt * 16 + xm0) * T_ * IN_ + xc0;
  const float* xp1 = x + (size_t)(bt * 16 + xm1) * T_ * IN_ + xc0;
  float xr0 = xp0[0];                    // t = 0
  float xr1 = xp1[0];

  // reduce lanes: same o for both halves, rows xm0 / xm1
  const int ro = tid & 127;
  const float biasr = bias[ro];
  float* outp0 = out + (size_t)(bt * 16 + xm0) * T_ * OUT_ + ro;
  float* outp1 = out + (size_t)(bt * 16 + xm1) * T_ * OUT_ + ro;

  v8f s0 = {0.f,0.f,0.f,0.f,0.f,0.f,0.f,0.f};
  v8f s1 = {0.f,0.f,0.f,0.f,0.f,0.f,0.f,0.f};
  const v8f vzero = {0.f,0.f,0.f,0.f,0.f,0.f,0.f,0.f};

  for (int t = 0; t < T_; ++t) {
    // Opaque zero per iteration: blocks LICM of weight loads (no hoist -> no spill).
    int tinv = 0;
    asm volatile("" : "+v"(tinv));

    // ---- stage prefetched x_t tile -> LDS (bf16) ----
    bufX[xm0][xc0] = (bf16t)xr0;
    bufX[xm1][xc0] = (bf16t)xr1;
    __syncthreads();

    // ---- phase 1: h = x @ t0 + s0  -> bufA ----
    v8f h = mm_phase<4>(&bufX[0][0], IN_ + XPAD, t0t, IN_, lane, w << 4, 0, tinv, s0);
#pragma unroll
    for (int r = 0; r < 8; ++r) bufA[mrow + r][ncol] = (bf16t)h[r];
    __syncthreads();

    // prefetch next step's x into registers (global latency hidden behind 4 phases)
    if (t + 1 < T_) {
      xr0 = xp0[(size_t)(t + 1) * IN_];
      xr1 = xp1[(size_t)(t + 1) * IN_];
    }

    // ---- phase 2: h = h @ r0 ; act*a0 -> new s0 ; -> bufB ----
    h = mm_phase<16>(&bufA[0][0], L_ + APAD, r0t, L_, lane, w << 4, 0, tinv, vzero);
#pragma unroll
    for (int r = 0; r < 8; ++r) {
      float v = mixed_act(h[r], grp) * alpha0;
      s0[r] = v;
      bufB[mrow + r][ncol] = (bf16t)v;
    }
    __syncthreads();

    // ---- phase 3: h = h @ t1 + s1 -> bufA ----
    h = mm_phase<16>(&bufB[0][0], L_ + APAD, t1t, L_, lane, w << 4, 0, tinv, s1);
#pragma unroll
    for (int r = 0; r < 8; ++r) bufA[mrow + r][ncol] = (bf16t)h[r];
    __syncthreads();

    // ---- phase 4: h = h @ r1 ; act*a1 -> new s1 ; -> bufB ----
    h = mm_phase<16>(&bufA[0][0], L_ + APAD, r1t, L_, lane, w << 4, 0, tinv, vzero);
#pragma unroll
    for (int r = 0; r < 8; ++r) {
      float v = mixed_act(h[r], grp) * alpha1;
      s1[r] = v;
      bufB[mrow + r][ncol] = (bf16t)v;
    }
    __syncthreads();

    // ---- readout: out = h @ W^T + b ; split-K over 4 wave groups ----
    {
      v8f rt = mm_phase<4>(&bufB[0][0], L_ + APAD, Wb, L_, lane, rntile << 4,
                           rkc * 128, tinv, vzero);
      int n = (rntile << 4) + (lane & 15);
#pragma unroll
      for (int r = 0; r < 8; ++r) part[rkc][mrow + r][n] = rt[r];
    }
    __syncthreads();

    // ---- reduce partials + bias; emit via async LDS->global stores ----
    {
      float v0 = part[0][xm0][ro] + part[1][xm0][ro] + part[2][xm0][ro]
               + part[3][xm0][ro] + biasr;
      float v1 = part[0][xm1][ro] + part[1][xm1][ro] + part[2][xm1][ro]
               + part[3][xm1][ro] + biasr;

      const int slot = t & 3;
      float* st0 = &outStage[slot][tid];
      float* st1 = &outStage[slot][1024 + tid];
      // ring pacing: each step issues 2 async stores; depth-4 ring -> oldest
      // slot's pair must have retired before we overwrite it (<=6 outstanding)
      asm volatile("s_wait_asynccnt 0x6" ::: "memory");
      *st0 = v0;
      *st1 = v1;
      // async engine reads LDS; DS and ASYNC queues are unordered -> drain DS first
      asm volatile("s_wait_dscnt 0x0" ::: "memory");
      async_store_f32(outp0 + (size_t)t * OUT_, st0);
      async_store_f32(outp1 + (size_t)t * OUT_, st1);
    }
    // NOTE: no barrier here. `part`/`outStage` are next written in step t+1's
    // readout/reduce, which is separated by 5 barriers; outStage reuse is paced
    // by s_wait_asynccnt. s_endpgm's implicit wait-idle drains the async tail.
  }
}

extern "C" void kernel_launch(void* const* d_in, const int* in_sizes, int n_in,
                              void* d_out, int out_size, void* d_ws, size_t ws_size,
                              hipStream_t stream) {
  const float* x  = (const float*)d_in[0];
  const float* t0 = (const float*)d_in[1];
  const float* t1 = (const float*)d_in[2];
  const float* r0 = (const float*)d_in[3];
  const float* r1 = (const float*)d_in[4];
  const float* a0 = (const float*)d_in[5];
  const float* a1 = (const float*)d_in[6];
  const float* W  = (const float*)d_in[7];
  const float* bb = (const float*)d_in[8];
  float* out = (float*)d_out;

  // workspace: bf16 weights, K contiguous per output column ([N][K])
  char* ws = (char*)d_ws;
  bf16t* t0t = (bf16t*)(ws + 0);          // 512*128*2 = 131072 B
  bf16t* r0t = (bf16t*)(ws + 131072);     // 512*512*2 = 524288 B
  bf16t* t1t = (bf16t*)(ws + 655360);
  bf16t* r1t = (bf16t*)(ws + 1179648);
  bf16t* Wb  = (bf16t*)(ws + 1703936);    // 128*512*2 = 131072 B  (total 1835008 B)

  transpose_to_bf16<<<(128 * 512 + 255) / 256, 256, 0, stream>>>(t0, t0t, 128, 512);
  transpose_to_bf16<<<(512 * 512 + 255) / 256, 256, 0, stream>>>(r0, r0t, 512, 512);
  transpose_to_bf16<<<(512 * 512 + 255) / 256, 256, 0, stream>>>(t1, t1t, 512, 512);
  transpose_to_bf16<<<(512 * 512 + 255) / 256, 256, 0, stream>>>(r1, r1t, 512, 512);
  convert_bf16<<<(128 * 512 + 255) / 256, 256, 0, stream>>>(W, Wb, 128 * 512);

  // 4 independent batch tiles (M=16 each), 32 waves/WGP, state held in registers
  reservoir_kernel<<<4, 1024, 0, stream>>>(x, a0, a1, bb, t0t, r0t, t1t, r1t, Wb, out);
}
